// GQA_FlashAttn_45457933861106
// MI455X (gfx1250) — matrix operand order repaired and checked by reading, not compiled, not run on hardware
//
#include <hip/hip_runtime.h>
#include <hip/hip_bf16.h>
#include <math.h>

#define SEQ    2048
#define D_IN   2048
#define NH     32
#define NKV    8
#define HD     64
#define GRP    (NH / NKV)   // 4

typedef __attribute__((ext_vector_type(16))) _Float16 v16h;
typedef __attribute__((ext_vector_type(8)))  float    v8f;
typedef __attribute__((ext_vector_type(4)))  unsigned u32x4;
typedef __attribute__((ext_vector_type(8)))  int      i32x8;
typedef __attribute__((ext_vector_type(4)))  int      i32x4;

union Frag {
    v16h  v;
    unsigned u[8];
    u32x4 q[2];
};

// ---------------------------------------------------------------------------
// fp32 -> f16 conversion (row-major preserved) — used for x
// ---------------------------------------------------------------------------
__global__ void cvt_f32_f16(const float* __restrict__ in, _Float16* __restrict__ out, int n) {
    int i = blockIdx.x * blockDim.x + threadIdx.x;
    if (i < n) out[i] = (_Float16)in[i];
}

// ---------------------------------------------------------------------------
// fp32 [R,C] -> f16 [C,R] tiled transpose-convert — used for weights
// block (32,8), grid (C/32, R/32)
// ---------------------------------------------------------------------------
__global__ __launch_bounds__(256)
void cvt_transpose(const float* __restrict__ in, _Float16* __restrict__ out, int R, int C) {
    __shared__ _Float16 tile[32][33];
    const int c0 = blockIdx.x * 32;
    const int r0 = blockIdx.y * 32;
    for (int i = threadIdx.y; i < 32; i += 8)
        tile[i][threadIdx.x] = (_Float16)in[(size_t)(r0 + i) * C + c0 + threadIdx.x];
    __syncthreads();
    for (int i = threadIdx.y; i < 32; i += 8)
        out[(size_t)(c0 + i) * R + r0 + threadIdx.x] = tile[threadIdx.x][i];
}

// V: [S, NKV*HD] fp32 -> [NKV, HD, S] f16 (head-major, dim-major = V^T per head)
__global__ void v_pack_t(const float* __restrict__ in, _Float16* __restrict__ out) {
    int i = blockIdx.x * blockDim.x + threadIdx.x;  // over SEQ*NKV*HD
    int s = i / (NKV * HD);
    int r = i % (NKV * HD);
    int h = r / HD;
    int d = r % HD;
    out[((size_t)h * HD + d) * SEQ + s] = (_Float16)in[i];
}

// ---------------------------------------------------------------------------
// WMMA GEMM: C[M,N] = A[M,K] * B[K,N]; A row-major f16, BT = B^T (N x K) f16.
// Block = 256 threads (8 waves), block tile 128x64, wave tile 16x64.
// B^T k-tile (64 rows x 32 cols f16 = 4KB) staged into LDS by the Tensor
// Data Mover (tensor_load_to_lds + s_wait_tensorcnt), fragments read as
// b128 DS loads. grid = (N/64, M/128).
// ---------------------------------------------------------------------------
__global__ __launch_bounds__(256)
void gemm_f16_wmma(const _Float16* __restrict__ A, const _Float16* __restrict__ BT,
                   float* __restrict__ C, int M, int N, int K) {
    __shared__ __align__(16) _Float16 Blds[64 * 32];

    const int wave = threadIdx.x >> 5;
    const int lane = threadIdx.x & 31;
    const int l15  = lane & 15;
    const int half = lane >> 4;

    const int m0 = blockIdx.y * 128 + wave * 16;
    const int n0 = blockIdx.x * 64;
    const int mrow = m0 + l15;

    v8f acc[4] = {v8f{}, v8f{}, v8f{}, v8f{}};

    const unsigned ldsoff = (unsigned)(uintptr_t)&Blds[0];

    for (int kb = 0; kb < K; kb += 32) {
        if (wave == 0) {
            // ---- TDM: 2D tile load BT[n0..n0+63][kb..kb+31] -> Blds[64][32]
            unsigned long long ga =
                (unsigned long long)(uintptr_t)(BT + (size_t)n0 * K + kb);
            u32x4 g0;
            g0[0] = 1u;                                   // count = 1 (valid D#)
            g0[1] = ldsoff;                               // lds_addr
            g0[2] = (unsigned)ga;                         // global_addr[31:0]
            g0[3] = (unsigned)((ga >> 32) & 0x1FFFFFFull) // global_addr[56:32]
                    | (2u << 30);                         // type = 2 (image)
            i32x8 g1;
            g1[0] = (int)(1u << 16);                      // data_size = 1 (2 bytes)
            g1[1] = (int)(((unsigned)K & 0xFFFFu) << 16); // tensor_dim0 lo16
            g1[2] = (int)(((unsigned)K >> 16) |
                          (((unsigned)N & 0xFFFFu) << 16)); // dim0 hi / dim1 lo
            g1[3] = (int)(((unsigned)N >> 16) | (32u << 16)); // dim1 hi | tile_dim0=32
            g1[4] = 64;                                   // tile_dim1 = 64
            g1[5] = (int)(unsigned)K;                     // tensor_dim0_stride
            g1[6] = 0;
            g1[7] = 0;
            i32x4 gz4 = i32x4{0, 0, 0, 0};
            i32x8 gz8 = i32x8{0, 0, 0, 0, 0, 0, 0, 0};
            __builtin_amdgcn_tensor_load_to_lds(g0, g1, gz4, gz4, gz8, 0);
            __builtin_amdgcn_s_wait_tensorcnt(0);
        }
        __syncthreads();

        // A fragment: two contiguous b128 loads per lane
        const _Float16* Arow = A + (size_t)mrow * K + kb;
        __builtin_prefetch((const void*)(Arow + 32), 0, 0);
        Frag a;
        a.q[0] = *(const u32x4*)(Arow + half * 8);
        a.q[1] = *(const u32x4*)(Arow + 16 + half * 8);

        // 4 B fragments from LDS: lane = N row, K = half*8 + 0..7 and 16 + half*8 + 0..7
#pragma unroll
        for (int t = 0; t < 4; ++t) {
            const _Float16* Brow = Blds + (t * 16 + l15) * 32 + half * 8;
            Frag b;
            b.q[0] = *(const u32x4*)(Brow);
            b.q[1] = *(const u32x4*)(Brow + 16);
            acc[t] = __builtin_amdgcn_wmma_f32_16x16x32_f16(
                false, a.v, false, b.v, (short)0, acc[t], false, false);
        }
        __syncthreads();
    }

#pragma unroll
    for (int t = 0; t < 4; ++t) {
        const int n = n0 + t * 16 + l15;
#pragma unroll
        for (int r = 0; r < 8; ++r) {
            const int m = m0 + r + 8 * half;
            C[(size_t)m * N + n] = acc[t][r];
        }
    }
}

// ---------------------------------------------------------------------------
// Per-head RMSNorm + RoPE; fp32 in [S, H*HD], f16 out [H, S, HD].
// One wave per (s, h). Lane handles dims d=lane and d=lane+32.
// ---------------------------------------------------------------------------
__global__ __launch_bounds__(32)
void rmsnorm_rope(const float* __restrict__ in, const float* __restrict__ w,
                  const float* __restrict__ cosb, const float* __restrict__ sinb,
                  _Float16* __restrict__ out, int H) {
    const int s = blockIdx.x / H;
    const int h = blockIdx.x % H;
    const int lane = threadIdx.x;

    const float* v = in + (size_t)s * (H * HD) + h * HD;
    float a0 = v[lane];
    float a1 = v[lane + 32];
    float ss = a0 * a0 + a1 * a1;
#pragma unroll
    for (int off = 1; off < 32; off <<= 1) ss += __shfl_xor(ss, off, 32);
    const float r = rsqrtf(ss * (1.0f / 64.0f) + 1e-6f);

    const float n0 = a0 * r * w[lane];
    const float n1 = a1 * r * w[lane + 32];

    const float c0 = cosb[s * HD + lane];
    const float s0 = sinb[s * HD + lane];
    const float c1 = cosb[s * HD + lane + 32];
    const float s1 = sinb[s * HD + lane + 32];

    const float o0 = n0 * c0 - n1 * s0;   // d < 32 : rot = -x[d+32]
    const float o1 = n1 * c1 + n0 * s1;   // d >= 32: rot =  x[d-32]

    _Float16* op = out + ((size_t)h * SEQ + s) * HD;
    op[lane]      = (_Float16)o0;
    op[lane + 32] = (_Float16)o1;
}

// ---------------------------------------------------------------------------
// Flash attention, online softmax. 4 waves / block, one 16-query tile each.
// Q [NH,S,HD] f16, K [NKV,S,HD] f16, Vt [NKV,HD,S] f16, out [S, NH*HD] f16.
// grid = (SEQ/64, NH), block = 128.
// ---------------------------------------------------------------------------
__global__ __launch_bounds__(128)
void flash_attn(const _Float16* __restrict__ Qr, const _Float16* __restrict__ Kr,
                const _Float16* __restrict__ Vt, _Float16* __restrict__ Oa) {
    __shared__ __align__(16) _Float16 Plds[4][16 * 32];

    const int wave = threadIdx.x >> 5;
    const int lane = threadIdx.x & 31;
    const int l15  = lane & 15;
    const int half = lane >> 4;
    const int h    = blockIdx.y;
    const int kvh  = h / GRP;
    const int q0   = (blockIdx.x * 4 + wave) * 16;

    _Float16* Pl = Plds[wave];
    const _Float16* Qh = Qr + (size_t)h * SEQ * HD;
    const _Float16* Kh = Kr + (size_t)kvh * SEQ * HD;
    const _Float16* Vh = Vt + (size_t)kvh * HD * SEQ;

    // Q A-fragments: two K-chunks (dims 0..31, 32..63), b128 loads
    Frag qa[2];
    {
        const _Float16* Qrow = Qh + (size_t)(q0 + l15) * HD;
#pragma unroll
        for (int c = 0; c < 2; ++c) {
            qa[c].q[0] = *(const u32x4*)(Qrow + c * 32 + half * 8);
            qa[c].q[1] = *(const u32x4*)(Qrow + c * 32 + 16 + half * 8);
        }
    }

    v8f o[4] = {v8f{}, v8f{}, v8f{}, v8f{}};
    float mrow[8], lrow[8];
#pragma unroll
    for (int r = 0; r < 8; ++r) { mrow[r] = -3.0e38f; lrow[r] = 0.0f; }

    const int kt_max = (q0 + 15) / 32;
    for (int kt = 0; kt <= kt_max; ++kt) {
        const int k0 = kt * 32;

        // S(16x32) = Q(16x64) @ K^T(64x32): two 16x16 C-tiles, 2 WMMAs each
        v8f sa[2] = {v8f{}, v8f{}};
#pragma unroll
        for (int t = 0; t < 2; ++t) {
            const _Float16* Krow = Kh + (size_t)(k0 + t * 16 + l15) * HD;
#pragma unroll
            for (int c = 0; c < 2; ++c) {
                Frag b;
                b.q[0] = *(const u32x4*)(Krow + c * 32 + half * 8);
                b.q[1] = *(const u32x4*)(Krow + c * 32 + 16 + half * 8);
                sa[t] = __builtin_amdgcn_wmma_f32_16x16x32_f16(
                    false, qa[c].v, false, b.v, (short)0, sa[t], false, false);
            }
        }

        // scale + causal mask (C layout: row = r+8*half, key = k0 + t*16 + l15)
#pragma unroll
        for (int t = 0; t < 2; ++t) {
            const int key = k0 + t * 16 + l15;
#pragma unroll
            for (int r = 0; r < 8; ++r) {
                const int m = q0 + r + 8 * half;
                float v = sa[t][r] * 0.125f;
                sa[t][r] = (key > m) ? -1.0e30f : v;
            }
        }

        // row-max over the 16 lanes sharing each row
        float pmax[8];
#pragma unroll
        for (int r = 0; r < 8; ++r) {
            float v = fmaxf(sa[0][r], sa[1][r]);
#pragma unroll
            for (int off = 1; off < 16; off <<= 1) v = fmaxf(v, __shfl_xor(v, off, 32));
            pmax[r] = v;
        }

        // online-softmax rescale
        float rs[8];
#pragma unroll
        for (int r = 0; r < 8; ++r) {
            const float mnew = fmaxf(mrow[r], pmax[r]);
            const float corr = __expf(mrow[r] - mnew);
            mrow[r] = mnew;
            lrow[r] *= corr;
            rs[r] = corr;
        }
#pragma unroll
        for (int t = 0; t < 4; ++t)
#pragma unroll
            for (int r = 0; r < 8; ++r) o[t][r] *= rs[r];

        // P = exp(S - m), row sums
#pragma unroll
        for (int t = 0; t < 2; ++t)
#pragma unroll
            for (int r = 0; r < 8; ++r) sa[t][r] = __expf(sa[t][r] - mrow[r]);
#pragma unroll
        for (int r = 0; r < 8; ++r) {
            float v = sa[0][r] + sa[1][r];
#pragma unroll
            for (int off = 1; off < 16; off <<= 1) v += __shfl_xor(v, off, 32);
            lrow[r] += v;
        }

        // P (C layout) -> per-wave LDS -> A-fragment (16x32 f16).
        // Same-wave RAW through LDS: compiler inserts s_wait_dscnt.
#pragma unroll
        for (int t = 0; t < 2; ++t)
#pragma unroll
            for (int r = 0; r < 8; ++r)
                Pl[(r + 8 * half) * 32 + t * 16 + l15] = (_Float16)sa[t][r];
        Frag pa;
        {
            const _Float16* Prow = Pl + l15 * 32;
            pa.q[0] = *(const u32x4*)(Prow + half * 8);
            pa.q[1] = *(const u32x4*)(Prow + 16 + half * 8);
        }

        // O(16x64) += P(16x32) @ V(32x64); V^T rows give contiguous key pairs
#pragma unroll
        for (int t = 0; t < 4; ++t) {
            const _Float16* Vrow = Vh + (size_t)(t * 16 + l15) * SEQ + k0 + half * 8;
            Frag vb;
            vb.q[0] = *(const u32x4*)(Vrow);
            vb.q[1] = *(const u32x4*)(Vrow + 16);
            o[t] = __builtin_amdgcn_wmma_f32_16x16x32_f16(
                false, pa.v, false, vb.v, (short)0, o[t], false, false);
        }
    }

    // epilogue: divide by row sum, write [S, NH*HD] f16
#pragma unroll
    for (int r = 0; r < 8; ++r) {
        const float inv = 1.0f / lrow[r];
        const int m = q0 + r + 8 * half;
#pragma unroll
        for (int t = 0; t < 4; ++t) {
            const int d = t * 16 + l15;
            Oa[(size_t)m * (NH * HD) + h * HD + d] = (_Float16)(o[t][r] * inv);
        }
    }
}

// ---------------------------------------------------------------------------
// launch
// ---------------------------------------------------------------------------
extern "C" void kernel_launch(void* const* d_in, const int* in_sizes, int n_in,
                              void* d_out, int out_size, void* d_ws, size_t ws_size,
                              hipStream_t stream) {
    (void)in_sizes; (void)n_in; (void)out_size; (void)ws_size;

    const float* x    = (const float*)d_in[0];
    const float* cosb = (const float*)d_in[1];
    const float* sinb = (const float*)d_in[2];
    const float* wq   = (const float*)d_in[3];
    const float* wk   = (const float*)d_in[4];
    const float* wv   = (const float*)d_in[5];
    const float* wo   = (const float*)d_in[6];
    const float* qnw  = (const float*)d_in[7];
    const float* knw  = (const float*)d_in[8];
    float* out = (float*)d_out;

    char* ws = (char*)d_ws;
    const size_t MB = 1u << 20;
    _Float16* xh   = (_Float16*)(ws + 0 * MB);    // 8 MB   x f16 [S, D_IN]
    _Float16* wqT  = (_Float16*)(ws + 8 * MB);    // 8 MB   wq^T [2048, D_IN]
    _Float16* wkT  = (_Float16*)(ws + 16 * MB);   // 2 MB   wk^T [512, D_IN]
    _Float16* wvT  = (_Float16*)(ws + 18 * MB);   // 2 MB   wv^T [512, D_IN]
    _Float16* woT  = (_Float16*)(ws + 20 * MB);   // 8 MB   wo^T [2048, 2048]
    float*    Qf   = (float*)   (ws + 28 * MB);   // 16 MB
    float*    Kf   = (float*)   (ws + 44 * MB);   // 4 MB
    float*    Vf   = (float*)   (ws + 48 * MB);   // 4 MB
    _Float16* Qh   = (_Float16*)(ws + 52 * MB);   // 8 MB   [NH, S, HD]
    _Float16* Kh   = (_Float16*)(ws + 60 * MB);   // 2 MB   [NKV, S, HD]
    _Float16* VhT  = (_Float16*)(ws + 62 * MB);   // 2 MB   [NKV, HD, S]
    _Float16* Ah   = (_Float16*)(ws + 64 * MB);   // 8 MB   attn f16 [S, NH*HD]

    // 1) conversions: x -> f16; weights -> transposed f16
    {
        int n = SEQ * D_IN;
        cvt_f32_f16<<<(n + 255) / 256, 256, 0, stream>>>(x, xh, n);
    }
    cvt_transpose<<<dim3((NH * HD) / 32, D_IN / 32), dim3(32, 8), 0, stream>>>(
        wq, wqT, D_IN, NH * HD);
    cvt_transpose<<<dim3((NKV * HD) / 32, D_IN / 32), dim3(32, 8), 0, stream>>>(
        wk, wkT, D_IN, NKV * HD);
    cvt_transpose<<<dim3((NKV * HD) / 32, D_IN / 32), dim3(32, 8), 0, stream>>>(
        wv, wvT, D_IN, NKV * HD);
    cvt_transpose<<<dim3(D_IN / 32, (NH * HD) / 32), dim3(32, 8), 0, stream>>>(
        wo, woT, NH * HD, D_IN);

    // 2) QKV projections (WMMA GEMMs, TDM-staged B tiles)
    gemm_f16_wmma<<<dim3((NH * HD) / 64, SEQ / 128), 256, 0, stream>>>(
        xh, wqT, Qf, SEQ, NH * HD, D_IN);
    gemm_f16_wmma<<<dim3((NKV * HD) / 64, SEQ / 128), 256, 0, stream>>>(
        xh, wkT, Kf, SEQ, NKV * HD, D_IN);
    gemm_f16_wmma<<<dim3((NKV * HD) / 64, SEQ / 128), 256, 0, stream>>>(
        xh, wvT, Vf, SEQ, NKV * HD, D_IN);

    // 3) RMSNorm + RoPE (Q, K); V repack to [NKV, HD, S] f16
    rmsnorm_rope<<<SEQ * NH, 32, 0, stream>>>(Qf, qnw, cosb, sinb, Qh, NH);
    rmsnorm_rope<<<SEQ * NKV, 32, 0, stream>>>(Kf, knw, cosb, sinb, Kh, NKV);
    v_pack_t<<<(SEQ * NKV * HD) / 256, 256, 0, stream>>>(Vf, VhT);

    // 4) causal flash attention
    flash_attn<<<dim3(SEQ / 64, NH), 128, 0, stream>>>(Qh, Kh, VhT, Ah);

    // 5) output projection
    gemm_f16_wmma<<<dim3(D_IN / 64, SEQ / 128), 256, 0, stream>>>(
        Ah, woT, out, SEQ, D_IN, NH * HD);
}
